// DNA_Encoder_59425167507608
// MI455X (gfx1250) — matrix operand order repaired and checked by reading, not compiled, not run on hardware
//
#include <hip/hip_runtime.h>
#include <hip/hip_bf16.h>
#include <math.h>

#define N_NODES 30000
#define N_EDGES 300000
#define IN_DIM 256
#define HIDDEN 128
#define NUM_CLASSES 64
#define BATCH 10000
#define TOPK 16
#define NW 8   // waves per block in sim kernel

typedef __bf16 bf16_t;
typedef __attribute__((ext_vector_type(16))) __bf16 v16bf;
typedef __attribute__((ext_vector_type(8)))  __bf16 v8bf;
typedef __attribute__((ext_vector_type(8)))  float  v8f;

__device__ __forceinline__ bf16_t f2bf(float f) {
  unsigned u = __builtin_bit_cast(unsigned, f);
  unsigned r = (u + 0x7FFFu + ((u >> 16) & 1u)) >> 16;
  return __builtin_bit_cast(bf16_t, (unsigned short)r);
}

// ---------------- workspace layout ----------------
constexpr size_t alup(size_t x) { return (x + 255) & ~(size_t)255; }
constexpr size_t OFF_XBF = 0;                                                   // x as bf16 (N x 256)
constexpr size_t OFF_W1T = OFF_XBF + alup((size_t)N_NODES * IN_DIM * 2);        // W1^T bf16 (128 x 256)
constexpr size_t OFF_H   = OFF_W1T + alup((size_t)HIDDEN * IN_DIM * 2);         // h f32 (N x 128)
constexpr size_t OFF_T   = OFF_H   + alup((size_t)N_NODES * HIDDEN * 4);        // scatter accum (N x 128)
constexpr size_t OFF_Z   = OFF_T   + alup((size_t)N_NODES * HIDDEN * 4);        // z f32 (N x 128)
constexpr size_t OFF_DEG = OFF_Z   + alup((size_t)N_NODES * HIDDEN * 4);
constexpr size_t OFF_DIS = OFF_DEG + alup((size_t)N_NODES * 4);
constexpr size_t OFF_CNT = OFF_DIS + alup((size_t)N_NODES * 4);
constexpr size_t OFF_ENB = OFF_CNT + alup((size_t)N_NODES * 4);                 // en bf16 (B x 128)
constexpr size_t OFF_PLC = OFF_ENB + alup((size_t)BATCH * HIDDEN * 2);          // p_lc (B x 64)
constexpr size_t OFF_TV  = OFF_PLC + alup((size_t)BATCH * NUM_CLASSES * 4);     // top-k vals
constexpr size_t OFF_TI  = OFF_TV  + alup((size_t)BATCH * TOPK * 4);            // top-k idx

// ---------------- conversion kernels ----------------
__global__ void cvt_x_kernel(const float* __restrict__ x, bf16_t* __restrict__ xb) {
  size_t tid = (size_t)blockIdx.x * blockDim.x + threadIdx.x;
  if (tid < (size_t)N_NODES * IN_DIM) xb[tid] = f2bf(x[tid]);
}

__global__ void cvt_w1t_kernel(const float* __restrict__ w1, bf16_t* __restrict__ w1t) {
  int tid = blockIdx.x * blockDim.x + threadIdx.x;
  if (tid < HIDDEN * IN_DIM) {
    int n = tid / IN_DIM, k = tid % IN_DIM;          // w1t[n][k] = w1[k][n]
    w1t[tid] = f2bf(w1[(size_t)k * HIDDEN + n]);
  }
}

__global__ void init_kernel(float* __restrict__ t, float* __restrict__ deg, float* __restrict__ cnt) {
  size_t tid = (size_t)blockIdx.x * blockDim.x + threadIdx.x;
  if (tid < (size_t)N_NODES * HIDDEN) t[tid] = 0.0f;
  if (tid < N_NODES) { deg[tid] = 1.0f; cnt[tid] = 0.0f; }   // deg starts at 1 (self loop)
}

// ---------------- GEMM1: h = relu(x @ W1 + b1), WMMA bf16 ----------------
__global__ void gemm1_kernel(const bf16_t* __restrict__ xb, const bf16_t* __restrict__ w1t,
                             const float* __restrict__ b1, float* __restrict__ h) {
  const int mt = blockIdx.x, nt = blockIdx.y;
  const int l = threadIdx.x, lm = l & 15, hi = l >> 4;
  const bf16_t* arow = xb  + (size_t)(mt * 16 + lm) * IN_DIM;
  const bf16_t* brow = w1t + (size_t)(nt * 16 + lm) * IN_DIM;
  v8f acc = {};
  #pragma unroll
  for (int k0 = 0; k0 < IN_DIM; k0 += 32) {
    v16bf A = *(const v16bf*)(arow + k0 + hi * 16);
    v16bf B = *(const v16bf*)(brow + k0 + hi * 16);
    acc = __builtin_amdgcn_wmma_f32_16x16x32_bf16(false, A, false, B, (short)0, acc, false, false);
  }
  const int nn = nt * 16 + lm;
  const float bias = b1[nn];
  #pragma unroll
  for (int v = 0; v < 8; v++) {
    int mm = mt * 16 + v + 8 * hi;
    h[(size_t)mm * HIDDEN + nn] = fmaxf(acc[v] + bias, 0.0f);
  }
}

// ---------------- graph side ----------------
__global__ void deg_kernel(const int* __restrict__ col, float* __restrict__ deg) {
  int e = blockIdx.x * blockDim.x + threadIdx.x;
  if (e < N_EDGES) unsafeAtomicAdd(&deg[col[e]], 1.0f);
}

__global__ void dis_kernel(const float* __restrict__ deg, float* __restrict__ dis) {
  int n = blockIdx.x * blockDim.x + threadIdx.x;
  if (n < N_NODES) dis[n] = rsqrtf(deg[n]);          // deg >= 1 always
}

__global__ void scatter_kernel(const int* __restrict__ row, const int* __restrict__ col,
                               const float* __restrict__ dis, const float* __restrict__ h,
                               float* __restrict__ t, float* __restrict__ cnt) {
  size_t tid = (size_t)blockIdx.x * blockDim.x + threadIdx.x;
  if (tid >= (size_t)N_EDGES * HIDDEN) return;
  int e = (int)(tid >> 7), f = (int)(tid & 127);
  int r = row[e], c = col[e];
  float dr = dis[r];
  unsafeAtomicAdd(&t[(size_t)c * HIDDEN + f], dr * h[(size_t)r * HIDDEN + f]);
  if (f == 0) unsafeAtomicAdd(&cnt[c], dr);
}

// z[n] = relu( (dis[n]*(t[n]+dis[n]*h[n])) @grouped v_w  +  dis[n]*(cnt[n]+dis[n]) * v_b )
__global__ void node_v_kernel(const float* __restrict__ t, const float* __restrict__ h,
                              const float* __restrict__ dis, const float* __restrict__ cnt,
                              const float* __restrict__ vw, const float* __restrict__ vb,
                              float* __restrict__ z) {
  size_t tid = (size_t)blockIdx.x * blockDim.x + threadIdx.x;
  if (tid >= (size_t)N_NODES * HIDDEN) return;
  int n = (int)(tid >> 7), o = (int)(tid & 127);
  int g = o >> 4, oo = o & 15;
  float d = dis[n];
  float acc = d * (cnt[n] + d) * vb[o];
  const float* tr = t + (size_t)n * HIDDEN + g * 16;
  const float* hr = h + (size_t)n * HIDDEN + g * 16;
  const float* w  = vw + g * 256 + oo;               // v_w[g][i][oo]
  #pragma unroll
  for (int i = 0; i < 16; i++) {
    float s = d * (tr[i] + d * hr[i]);
    acc = fmaf(s, w[i * 16], acc);
  }
  z[tid] = fmaxf(acc, 0.0f);
}

// ---------------- heads ----------------
__global__ void norm_embed_kernel(const float* __restrict__ z, float* __restrict__ emb_out,
                                  bf16_t* __restrict__ enb) {
  __shared__ float red[128];
  int b = blockIdx.x, t = threadIdx.x;
  float v = z[(size_t)b * HIDDEN + t];
  emb_out[(size_t)b * HIDDEN + t] = v;
  red[t] = v * v;
  __syncthreads();
  for (int s = 64; s > 0; s >>= 1) { if (t < s) red[t] += red[t + s]; __syncthreads(); }
  float inv = 1.0f / fmaxf(sqrtf(red[0]), 1e-8f);
  enb[(size_t)b * HIDDEN + t] = f2bf(v * inv);
}

__global__ void lin2_kernel(const float* __restrict__ z, const float* __restrict__ w2,
                            const float* __restrict__ b2, float* __restrict__ plc) {
  __shared__ float red[64];
  int b = blockIdx.x, c = threadIdx.x;
  const float* zr = z + (size_t)b * HIDDEN;
  float acc = b2[c];
  #pragma unroll 8
  for (int k = 0; k < HIDDEN; k++) acc = fmaf(zr[k], w2[(size_t)k * NUM_CLASSES + c], acc);
  red[c] = acc; __syncthreads();
  for (int s = 32; s > 0; s >>= 1) { if (c < s) red[c] = fmaxf(red[c], red[c + s]); __syncthreads(); }
  float m = red[0]; __syncthreads();
  red[c] = expf(acc - m); __syncthreads();
  for (int s = 32; s > 0; s >>= 1) { if (c < s) red[c] += red[c + s]; __syncthreads(); }
  plc[(size_t)b * NUM_CLASSES + c] = acc - m - logf(red[0]);
}

// ---------------- sim = en@en^T tiled via WMMA + online top-16 ----------------
// Block: 8 waves, 16 query rows. Column chunks (625 tiles of 16) strided across
// waves; each wave keeps a private sorted top-16 per row in its LDS slice, then
// wave-0 lanes merge the 8 sorted lists. 625 blocks x 8 waves = 5000 waves.
__global__ void __launch_bounds__(32 * NW)
sim_topk_kernel(const bf16_t* __restrict__ enb,
                float* __restrict__ topv, int* __restrict__ topi) {
  __shared__ float ct[NW][16][16];
  __shared__ float tv[NW][16][16];
  __shared__ int   ti[NW][16][16];
  const int tid = threadIdx.x;
  const int w  = tid >> 5;       // wave id 0..7
  const int l  = tid & 31;       // lane
  const int lm = l & 15, hi = l >> 4;
  const int qb = blockIdx.x * 16;

  for (int i = l; i < 256; i += 32) { (&tv[w][0][0])[i] = -3.0e38f; (&ti[w][0][0])[i] = 0; }

  const bf16_t* qrow = enb + (size_t)(qb + lm) * HIDDEN;
  auto loadA = [&](int k0) -> v16bf {
    return *(const v16bf*)(qrow + k0 + hi * 16);
  };
  v16bf A0 = loadA(0), A1 = loadA(32), A2 = loadA(64), A3 = loadA(96);
  __syncthreads();

  const int NCH  = BATCH / 16;               // 625
  const int ITER = (NCH + NW - 1) / NW;      // 79 (uniform across waves for barriers)
  for (int it = 0; it < ITER; ++it) {
    const int cb = w + it * NW;
    const bool active = cb < NCH;
    if (active) {
      const bf16_t* crow = enb + (size_t)(cb * 16 + lm) * HIDDEN;
      v16bf B0 = *(const v16bf*)(crow + 0  + hi * 16);
      v16bf B1 = *(const v16bf*)(crow + 32 + hi * 16);
      v16bf B2 = *(const v16bf*)(crow + 64 + hi * 16);
      v16bf B3 = *(const v16bf*)(crow + 96 + hi * 16);
      v8f acc = {};
      acc = __builtin_amdgcn_wmma_f32_16x16x32_bf16(false, A0, false, B0, (short)0, acc, false, false);
      acc = __builtin_amdgcn_wmma_f32_16x16x32_bf16(false, A1, false, B1, (short)0, acc, false, false);
      acc = __builtin_amdgcn_wmma_f32_16x16x32_bf16(false, A2, false, B2, (short)0, acc, false, false);
      acc = __builtin_amdgcn_wmma_f32_16x16x32_bf16(false, A3, false, B3, (short)0, acc, false, false);
      #pragma unroll
      for (int v = 0; v < 8; v++) ct[w][v + 8 * hi][lm] = acc[v];
    }
    __syncthreads();   // order ct writes before reads (uniform)
    if (active && l < 16) {
      #pragma unroll
      for (int j = 0; j < 16; j++) {
        float val = ct[w][l][j];
        if (val > tv[w][l][15]) {              // keep descending; earlier idx wins ties
          int p = 15;
          while (p > 0 && tv[w][l][p - 1] < val) {
            tv[w][l][p] = tv[w][l][p - 1]; ti[w][l][p] = ti[w][l][p - 1]; --p;
          }
          tv[w][l][p] = val; ti[w][l][p] = cb * 16 + j;
        }
      }
    }
    __syncthreads();   // (uniform)
  }

  // merge 8 sorted top-16 lists per row -> tv[0]
  if (tid < 16) {
    const int r = tid;
    for (int wv = 1; wv < NW; ++wv) {
      #pragma unroll 1
      for (int k = 0; k < 16; ++k) {
        float v = tv[wv][r][k];
        if (!(v > tv[0][r][15])) break;        // wv list sorted descending
        int id = ti[wv][r][k];
        int p = 15;
        while (p > 0 && (tv[0][r][p - 1] < v ||
                         (tv[0][r][p - 1] == v && ti[0][r][p - 1] > id))) {
          tv[0][r][p] = tv[0][r][p - 1]; ti[0][r][p] = ti[0][r][p - 1]; --p;
        }
        tv[0][r][p] = v; ti[0][r][p] = id;
      }
    }
    #pragma unroll
    for (int k = 0; k < TOPK; ++k) {
      topv[(size_t)(qb + r) * TOPK + k] = tv[0][r][k];
      topi[(size_t)(qb + r) * TOPK + k] = ti[0][r][k];
    }
  }
}

// ---------------- fuse + log_softmax + blend ----------------
__global__ void final_kernel(const float* __restrict__ plc, const float* __restrict__ topv,
                             const int* __restrict__ topi, const int* __restrict__ y,
                             float* __restrict__ out) {
  __shared__ float red[64];
  int b = blockIdx.x, c = threadIdx.x;
  float f = 0.0f;
  #pragma unroll
  for (int k = 0; k < TOPK; k++) {
    int cls = y[topi[(size_t)b * TOPK + k]];
    float w = expf(topv[(size_t)b * TOPK + k]);
    f += (cls == c) ? w : 0.0f;
  }
  red[c] = f; __syncthreads();
  for (int s = 32; s > 0; s >>= 1) { if (c < s) red[c] = fmaxf(red[c], red[c + s]); __syncthreads(); }
  float m = red[0]; __syncthreads();
  red[c] = expf(f - m); __syncthreads();
  for (int s = 32; s > 0; s >>= 1) { if (c < s) red[c] += red[c + s]; __syncthreads(); }
  float psim = f - m - logf(red[0]);
  out[(size_t)b * NUM_CLASSES + c] = 0.5f * plc[(size_t)b * NUM_CLASSES + c] + 0.5f * psim;
}

// ---------------- launch ----------------
extern "C" void kernel_launch(void* const* d_in, const int* in_sizes, int n_in,
                              void* d_out, int out_size, void* d_ws, size_t ws_size,
                              hipStream_t stream) {
  const float* x  = (const float*)d_in[0];
  const float* w1 = (const float*)d_in[1];
  const float* b1 = (const float*)d_in[2];
  const float* vw = (const float*)d_in[7];
  const float* vb = (const float*)d_in[8];
  const float* w2 = (const float*)d_in[9];
  const float* b2 = (const float*)d_in[10];
  const int* ei   = (const int*)d_in[11];
  const int* y    = (const int*)d_in[12];
  const int* row = ei;
  const int* col = ei + N_EDGES;

  char* ws = (char*)d_ws;
  bf16_t* xb   = (bf16_t*)(ws + OFF_XBF);
  bf16_t* w1t  = (bf16_t*)(ws + OFF_W1T);
  float*  h    = (float*)(ws + OFF_H);
  float*  t    = (float*)(ws + OFF_T);
  float*  z    = (float*)(ws + OFF_Z);
  float*  deg  = (float*)(ws + OFF_DEG);
  float*  dis  = (float*)(ws + OFF_DIS);
  float*  cnt  = (float*)(ws + OFF_CNT);
  bf16_t* enb  = (bf16_t*)(ws + OFF_ENB);
  float*  plc  = (float*)(ws + OFF_PLC);
  float*  topv = (float*)(ws + OFF_TV);
  int*    topi = (int*)(ws + OFF_TI);

  float* out_final = (float*)d_out;                                // (B, 64)
  float* out_emb   = (float*)d_out + (size_t)BATCH * NUM_CLASSES;  // (B, 128)

  cvt_x_kernel<<<(N_NODES * IN_DIM + 255) / 256, 256, 0, stream>>>(x, xb);
  cvt_w1t_kernel<<<(HIDDEN * IN_DIM + 255) / 256, 256, 0, stream>>>(w1, w1t);
  init_kernel<<<(N_NODES * HIDDEN + 255) / 256, 256, 0, stream>>>(t, deg, cnt);
  gemm1_kernel<<<dim3(N_NODES / 16, HIDDEN / 16), 32, 0, stream>>>(xb, w1t, b1, h);
  deg_kernel<<<(N_EDGES + 255) / 256, 256, 0, stream>>>(col, deg);
  dis_kernel<<<(N_NODES + 255) / 256, 256, 0, stream>>>(deg, dis);
  scatter_kernel<<<(int)(((size_t)N_EDGES * HIDDEN + 255) / 256), 256, 0, stream>>>(row, col, dis, h, t, cnt);
  node_v_kernel<<<(N_NODES * HIDDEN + 255) / 256, 256, 0, stream>>>(t, h, dis, cnt, vw, vb, z);
  norm_embed_kernel<<<BATCH, 128, 0, stream>>>(z, out_emb, enb);
  lin2_kernel<<<BATCH, 64, 0, stream>>>(z, w2, b2, plc);
  sim_topk_kernel<<<BATCH / 16, 32 * NW, 0, stream>>>(enb, topv, topi);
  final_kernel<<<BATCH, 64, 0, stream>>>(plc, topv, topi, y, out_final);
}
